// LatentSDE_1219770712318
// MI455X (gfx1250) — compile-verified, matrix operand order checked by reading, not run on hardware
//
#include <hip/hip_runtime.h>

typedef __attribute__((ext_vector_type(16))) _Float16 v16h;
typedef __attribute__((ext_vector_type(8)))  float    v8f;

#define T_N 256
#define B_N 2048
#define OBS_N 50
#define HID_N 30
#define LAT_N 10
#define CTX_N 3
#define CLAMP_V 9.0f
#define LOG2PI_F 1.8378770664093453f

#define WMMA_F16F32(a,b,c) __builtin_amdgcn_wmma_f32_16x16x32_f16(false,(a),false,(b),(short)0,(c),false,false)

__device__ __forceinline__ void lds_fence() {
  asm volatile("s_wait_dscnt 0x0" ::: "memory");
}

__device__ __forceinline__ v8f vzero8() {
  v8f z = {0.f,0.f,0.f,0.f,0.f,0.f,0.f,0.f};
  return z;
}

// Gather a 16xK(=32 chunk) operand tile (A: MxK rows, B: NxK rows, both
// row-major in LDS) into the CDNA5 WMMA f16 operand layout:
// lane<16: row=lane, K = {k0..k0+7, k0+16..k0+23} with k0=0
// lane>=16: row=lane-16, k0=8
__device__ __forceinline__ v16h load_op(const _Float16* buf, int stride, int lane) {
  int row = lane & 15;
  int k0  = (lane & 16) ? 8 : 0;
  const _Float16* p = buf + row * stride + k0;
  v16h r;
  #pragma unroll
  for (int i = 0; i < 8; ++i) r[i] = p[i];
  #pragma unroll
  for (int i = 0; i < 8; ++i) r[8 + i] = p[16 + i];
  return r;
}

// C/D tile: lane holds col N=lane%16, rows r0..r0+7 with r0 = 8*(lane/16)
__device__ __forceinline__ void store_c_f16(v8f c, _Float16* buf, int stride, int lane) {
  int col = lane & 15;
  int r0  = (lane & 16) ? 8 : 0;
  #pragma unroll
  for (int i = 0; i < 8; ++i) buf[(r0 + i) * stride + col] = (_Float16)c[i];
}

__device__ __forceinline__ void store_c_f32(v8f c, float* buf, int stride, int lane) {
  int col = lane & 15;
  int r0  = (lane & 16) ? 8 : 0;
  #pragma unroll
  for (int i = 0; i < 8; ++i) buf[(r0 + i) * stride + col] = c[i];
}

__device__ __forceinline__ void store_c_f32_n(v8f c, float* buf, int stride, int ncols, int lane) {
  int col = lane & 15;
  int r0  = (lane & 16) ? 8 : 0;
  if (col < ncols) {
    #pragma unroll
    for (int i = 0; i < 8; ++i) buf[(r0 + i) * stride + col] = c[i];
  }
}

__device__ __forceinline__ float sigmoid_f(float x) { return 1.f / (1.f + __expf(-x)); }
__device__ __forceinline__ float softplus_f(float x) { return (x > 20.f) ? x : log1pf(__expf(x)); }
__device__ __forceinline__ float clamp_f(float x) { return fminf(fmaxf(x, -CLAMP_V), CLAMP_V); }

__device__ __forceinline__ void stage_w(_Float16* dst, const float* src,
                                        int R, int C, int dR, int dC, int tid, int nthr) {
  for (int i = tid; i < dR * dC; i += nthr) {
    int r = i / dC, c = i % dC;
    dst[i] = (_Float16)((r < R && c < C) ? src[r * C + c] : 0.f);
  }
}

__device__ __forceinline__ void stage_b(float* dst, const float* src, int N, int dN, int tid, int nthr) {
  for (int i = tid; i < dN; i += nthr) dst[i] = (i < N) ? src[i] : 0.f;
}

// ---------------------------------------------------------------- encoder ---

struct EncLds {
  _Float16 Wih[96 * 64];   // GRU Wih, gate-padded: rows g*32+j (j<30), cols<50
  _Float16 Whh[96 * 32];
  _Float16 W1[64 * 32];    // drift layer1 (64,30) -> 64x32
  _Float16 W2[32 * 64];    // drift layer2 (30,64) -> 32x64
  _Float16 CtxW[16 * 32];  // (3,30)  -> 16x32
  _Float16 Qz0W[32 * 32];  // (20,30) -> 32x32
  float bih[96], bhh[96], b1[64], b2[32], ctxb[16], qz0b[32];
  float tsb[T_N];
  _Float16 xbuf[4][16 * 64];
  _Float16 abuf[4][16 * 32];
  _Float16 hbuf[4][16 * 64];
};

__device__ __forceinline__ void drift_eval(
    v8f in0, v8f in1, v8f& k0, v8f& k1,
    _Float16* ab, _Float16* hb, const EncLds& s, int lane)
{
  store_c_f16(in0, ab, 32, lane);
  store_c_f16(in1, ab + 16, 32, lane);
  lds_fence();
  v16h a = load_op(ab, 32, lane);
  #pragma unroll
  for (int ntl = 0; ntl < 4; ++ntl) {
    v16h b = load_op(s.W1 + ntl * 16 * 32, 32, lane);
    v8f acc = vzero8();
    acc = WMMA_F16F32(a, b, acc);
    float bb = s.b1[ntl * 16 + (lane & 15)];
    #pragma unroll
    for (int i = 0; i < 8; ++i) acc[i] = tanhf(acc[i] + bb);
    store_c_f16(acc, hb + ntl * 16, 64, lane);
  }
  lds_fence();
  v16h a0 = load_op(hb, 64, lane);
  v16h a1 = load_op(hb + 32, 64, lane);
  #pragma unroll
  for (int ntl = 0; ntl < 2; ++ntl) {
    v16h b0 = load_op(s.W2 + ntl * 16 * 64, 64, lane);
    v16h b1 = load_op(s.W2 + ntl * 16 * 64 + 32, 64, lane);
    v8f acc = vzero8();
    acc = WMMA_F16F32(a0, b0, acc);
    acc = WMMA_F16F32(a1, b1, acc);
    float bb = s.b2[ntl * 16 + (lane & 15)];
    #pragma unroll
    for (int i = 0; i < 8; ++i) acc[i] += bb;
    if (ntl == 0) k0 = acc; else k1 = acc;
  }
}

__global__ __launch_bounds__(128) void enc_kernel(
    const float* __restrict__ xs, const float* __restrict__ ts,
    const float* __restrict__ z0_noise,
    const float* gWih, const float* gWhh, const float* gbih, const float* gbhh,
    const float* dW1, const float* db1, const float* dW2, const float* db2,
    const float* cW, const float* cb, const float* qW, const float* qb,
    const float* pz_mean, const float* pz_logstd,
    float* __restrict__ ctx_ws, float* __restrict__ zs_ws, float* __restrict__ accum)
{
  __shared__ EncLds s;
  const int tid = threadIdx.x, nthr = blockDim.x;

  for (int i = tid; i < 96 * 64; i += nthr) {
    int r = i >> 6, c = i & 63;
    int g = r >> 5, j = r & 31;
    s.Wih[i] = (_Float16)((j < HID_N && c < OBS_N) ? gWih[(g * HID_N + j) * OBS_N + c] : 0.f);
  }
  for (int i = tid; i < 96 * 32; i += nthr) {
    int r = i >> 5, c = i & 31;
    int g = r >> 5, j = r & 31;
    s.Whh[i] = (_Float16)((j < HID_N && c < HID_N) ? gWhh[(g * HID_N + j) * HID_N + c] : 0.f);
  }
  for (int i = tid; i < 96; i += nthr) {
    int g = i >> 5, j = i & 31;
    s.bih[i] = (j < HID_N) ? gbih[g * HID_N + j] : 0.f;
    s.bhh[i] = (j < HID_N) ? gbhh[g * HID_N + j] : 0.f;
  }
  stage_w(s.W1,   dW1, 64, HID_N, 64, 32, tid, nthr);
  stage_w(s.W2,   dW2, HID_N, 64, 32, 64, tid, nthr);
  stage_w(s.CtxW, cW,  CTX_N, HID_N, 16, 32, tid, nthr);
  stage_w(s.Qz0W, qW,  2 * LAT_N, HID_N, 32, 32, tid, nthr);
  stage_b(s.b1, db1, 64, 64, tid, nthr);
  stage_b(s.b2, db2, HID_N, 32, tid, nthr);
  stage_b(s.ctxb, cb, CTX_N, 16, tid, nthr);
  stage_b(s.qz0b, qb, 2 * LAT_N, 32, tid, nthr);
  for (int i = tid; i < T_N; i += nthr) s.tsb[i] = ts[i];
  __syncthreads();

  const int lane = tid & 31, w = tid >> 5;
  const int row0 = (blockIdx.x * 4 + w) * 16;
  _Float16* xb = s.xbuf[w];
  _Float16* ab = s.abuf[w];
  _Float16* hb = s.hbuf[w];

  v8f h0 = vzero8(), h1 = vzero8();

  for (int step = 0; step < T_N; ++step) {
    const int trev = T_N - 1 - step;
    for (int e = lane; e < 16 * 64; e += 32) {
      int r = e >> 6, c = e & 63;
      float v = (c < OBS_N) ? xs[((size_t)trev * B_N + row0 + r) * OBS_N + c] : 0.f;
      xb[e] = (_Float16)v;
    }
    store_c_f16(h0, ab, 32, lane);
    store_c_f16(h1, ab + 16, 32, lane);
    lds_fence();

    // ---- GRU cell ----
    v16h ha  = load_op(ab, 32, lane);
    v16h xa0 = load_op(xb, 64, lane);
    v16h xa1 = load_op(xb + 32, 64, lane);
    v8f rg[2], zg[2];
    #pragma unroll
    for (int ntl = 0; ntl < 4; ++ntl) {      // tiles 0,1 = r gate; 2,3 = z gate
      v16h w0 = load_op(s.Wih + ntl * 16 * 64, 64, lane);
      v16h w1 = load_op(s.Wih + ntl * 16 * 64 + 32, 64, lane);
      v8f gi = vzero8();
      gi = WMMA_F16F32(xa0, w0, gi);
      gi = WMMA_F16F32(xa1, w1, gi);
      v16h wh = load_op(s.Whh + ntl * 16 * 32, 32, lane);
      v8f gh = vzero8();
      gh = WMMA_F16F32(ha, wh, gh);
      float bi = s.bih[ntl * 16 + (lane & 15)];
      float bh = s.bhh[ntl * 16 + (lane & 15)];
      v8f o;
      #pragma unroll
      for (int i = 0; i < 8; ++i) o[i] = sigmoid_f(gi[i] + bi + gh[i] + bh);
      if (ntl < 2) rg[ntl] = o; else zg[ntl - 2] = o;
    }
    #pragma unroll
    for (int st = 0; st < 2; ++st) {         // tiles 4,5 = n gate + h update
      int ntl = 4 + st;
      v16h w0 = load_op(s.Wih + ntl * 16 * 64, 64, lane);
      v16h w1 = load_op(s.Wih + ntl * 16 * 64 + 32, 64, lane);
      v8f gi = vzero8();
      gi = WMMA_F16F32(xa0, w0, gi);
      gi = WMMA_F16F32(xa1, w1, gi);
      v16h wh = load_op(s.Whh + ntl * 16 * 32, 32, lane);
      v8f gh = vzero8();
      gh = WMMA_F16F32(ha, wh, gh);
      float bi = s.bih[ntl * 16 + (lane & 15)];
      float bh = s.bhh[ntl * 16 + (lane & 15)];
      v8f* hp = st ? &h1 : &h0;
      #pragma unroll
      for (int i = 0; i < 8; ++i) {
        float nv = tanhf(gi[i] + bi + rg[st][i] * (gh[i] + bh));
        (*hp)[i] = (1.f - zg[st][i]) * nv + zg[st][i] * (*hp)[i];
      }
    }

    // ---- RK4 ODE solve (4 substeps, 4 drift evals each) ----
    float dtv = (step < T_N - 1) ? (s.tsb[T_N - 2 - step] - s.tsb[T_N - 1 - step]) : 0.f;
    float sdt = dtv * 0.25f;
    for (int sub = 0; sub < 4; ++sub) {
      v8f k1a, k1b, k2a, k2b, k3a, k3b, k4a, k4b, ta, tb;
      drift_eval(h0, h1, k1a, k1b, ab, hb, s, lane);
      #pragma unroll
      for (int i = 0; i < 8; ++i) { ta[i] = h0[i] + 0.5f * sdt * k1a[i]; tb[i] = h1[i] + 0.5f * sdt * k1b[i]; }
      drift_eval(ta, tb, k2a, k2b, ab, hb, s, lane);
      #pragma unroll
      for (int i = 0; i < 8; ++i) { ta[i] = h0[i] + 0.5f * sdt * k2a[i]; tb[i] = h1[i] + 0.5f * sdt * k2b[i]; }
      drift_eval(ta, tb, k3a, k3b, ab, hb, s, lane);
      #pragma unroll
      for (int i = 0; i < 8; ++i) { ta[i] = h0[i] + sdt * k3a[i]; tb[i] = h1[i] + sdt * k3b[i]; }
      drift_eval(ta, tb, k4a, k4b, ab, hb, s, lane);
      float c6 = sdt * (1.f / 6.f);
      #pragma unroll
      for (int i = 0; i < 8; ++i) {
        h0[i] += c6 * (k1a[i] + 2.f * k2a[i] + 2.f * k3a[i] + k4a[i]);
        h1[i] += c6 * (k1b[i] + 2.f * k2b[i] + 2.f * k3b[i] + k4b[i]);
      }
    }

    // ---- ctx projection of hidden[trev] ----
    store_c_f16(h0, ab, 32, lane);
    store_c_f16(h1, ab + 16, 32, lane);
    lds_fence();
    {
      v16h a = load_op(ab, 32, lane);
      v16h b = load_op(s.CtxW, 32, lane);
      v8f acc = vzero8();
      acc = WMMA_F16F32(a, b, acc);
      int col = lane & 15;
      int r0 = (lane & 16) ? 8 : 0;
      if (col < CTX_N) {
        float cbv = s.ctxb[col];
        #pragma unroll
        for (int i = 0; i < 8; ++i)
          ctx_ws[((size_t)trev * B_N + row0 + r0 + i) * CTX_N + col] = acc[i] + cbv;
      }
    }
  }

  // ---- qz0 head on hidden[0], z0 sample, KL ----
  store_c_f16(h0, ab, 32, lane);
  store_c_f16(h1, ab + 16, 32, lane);
  lds_fence();
  {
    v16h a = load_op(ab, 32, lane);
    float* qf = (float*)xb;  // reuse x staging as 16x32 f32
    #pragma unroll
    for (int ntl = 0; ntl < 2; ++ntl) {
      v16h b = load_op(s.Qz0W + ntl * 16 * 32, 32, lane);
      v8f acc = vzero8();
      acc = WMMA_F16F32(a, b, acc);
      float bb = s.qz0b[ntl * 16 + (lane & 15)];
      #pragma unroll
      for (int i = 0; i < 8; ++i) acc[i] += bb;
      store_c_f32(acc, qf + ntl * 16, 32, lane);
    }
    lds_fence();
    float kll = 0.f;
    for (int idx = lane; idx < 16 * LAT_N; idx += 32) {
      int rr = idx / LAT_N, l = idx % LAT_N;
      float mean = qf[rr * 32 + l];
      float lsc  = clamp_f(qf[rr * 32 + LAT_N + l]);
      float nz = z0_noise[(size_t)(row0 + rr) * LAT_N + l];
      zs_ws[(size_t)(row0 + rr) * LAT_N + l] = mean + __expf(lsc) * nz;
      float pls = clamp_f(pz_logstd[l]);
      float pm  = pz_mean[l];
      float vr = __expf(2.f * (lsc - pls));
      float mt = (mean - pm) * __expf(-pls); mt *= mt;
      kll += pls - lsc + 0.5f * (vr + mt) - 0.5f;
    }
    atomicAdd(&accum[0], kll);
  }
}

// -------------------------------------------------------------- SDE scan ---

struct SdeLds {
  _Float16 fW1[32 * 32];   // (30,13) -> 32x32
  _Float16 fW2[16 * 32];   // (10,30) -> 16x32
  _Float16 hW1[32 * 32];   // (30,10) -> 32x32 (K cols 10..31 zero)
  _Float16 hW2[16 * 32];
  float fb1[32], fb2[16], hb1[32], hb2[16];
  float gW1[LAT_N * HID_N], gb1[LAT_N * HID_N], gW2[LAT_N * HID_N], gb2[16];
  float tsb[T_N];
  _Float16 inbuf[4][16 * 32];
  _Float16 spbuf[4][16 * 32];
  float zbuf[4][16 * LAT_N];
  float fvbuf[4][16 * LAT_N];
  float hvbuf[4][16 * LAT_N];
};

__global__ __launch_bounds__(128) void sde_kernel(
    const float* __restrict__ ts, const float* __restrict__ bm,
    const float* fW1g, const float* fb1g, const float* fW2g, const float* fb2g,
    const float* hW1g, const float* hb1g, const float* hW2g, const float* hb2g,
    const float* gW1g, const float* gb1g, const float* gW2g, const float* gb2g,
    const float* __restrict__ ctx_ws, float* __restrict__ zs_ws, float* __restrict__ accum)
{
  __shared__ SdeLds s;
  const int tid = threadIdx.x, nthr = blockDim.x;
  stage_w(s.fW1, fW1g, HID_N, LAT_N + CTX_N, 32, 32, tid, nthr);
  stage_w(s.fW2, fW2g, LAT_N, HID_N, 16, 32, tid, nthr);
  stage_w(s.hW1, hW1g, HID_N, LAT_N, 32, 32, tid, nthr);
  stage_w(s.hW2, hW2g, LAT_N, HID_N, 16, 32, tid, nthr);
  stage_b(s.fb1, fb1g, HID_N, 32, tid, nthr);
  stage_b(s.fb2, fb2g, LAT_N, 16, tid, nthr);
  stage_b(s.hb1, hb1g, HID_N, 32, tid, nthr);
  stage_b(s.hb2, hb2g, LAT_N, 16, tid, nthr);
  for (int i = tid; i < LAT_N * HID_N; i += nthr) {
    s.gW1[i] = gW1g[i]; s.gb1[i] = gb1g[i]; s.gW2[i] = gW2g[i];
  }
  stage_b(s.gb2, gb2g, LAT_N, 16, tid, nthr);
  for (int i = tid; i < T_N; i += nthr) s.tsb[i] = ts[i];
  __syncthreads();

  const int lane = tid & 31, w = tid >> 5;
  const int row0 = (blockIdx.x * 4 + w) * 16;
  _Float16* inb = s.inbuf[w];
  _Float16* spb = s.spbuf[w];
  float* zb  = s.zbuf[w];
  float* fvb = s.fvbuf[w];
  float* hvb = s.hvbuf[w];

  for (int i = lane; i < 16 * LAT_N; i += 32) zb[i] = zs_ws[(size_t)row0 * LAT_N + i];
  lds_fence();

  float lrl = 0.f;
  for (int t = 0; t < T_N - 1; ++t) {
    float dtv = s.tsb[t + 1] - s.tsb[t];
    float sq = sqrtf(dtv);
    for (int e = lane; e < 16 * 32; e += 32) {
      int r = e >> 5, c = e & 31;
      float v = 0.f;
      if (c < LAT_N) v = zb[r * LAT_N + c];
      else if (c < LAT_N + CTX_N) v = ctx_ws[((size_t)t * B_N + row0 + r) * CTX_N + (c - LAT_N)];
      inb[e] = (_Float16)v;
    }
    lds_fence();
    v16h a = load_op(inb, 32, lane);
    // f net
    #pragma unroll
    for (int ntl = 0; ntl < 2; ++ntl) {
      v16h b = load_op(s.fW1 + ntl * 16 * 32, 32, lane);
      v8f acc = vzero8();
      acc = WMMA_F16F32(a, b, acc);
      float bb = s.fb1[ntl * 16 + (lane & 15)];
      #pragma unroll
      for (int i = 0; i < 8; ++i) acc[i] = softplus_f(acc[i] + bb);
      store_c_f16(acc, spb + ntl * 16, 32, lane);
    }
    lds_fence();
    {
      v16h a2 = load_op(spb, 32, lane);
      v16h b = load_op(s.fW2, 32, lane);
      v8f acc = vzero8();
      acc = WMMA_F16F32(a2, b, acc);
      float bb = s.fb2[lane & 15];
      #pragma unroll
      for (int i = 0; i < 8; ++i) acc[i] += bb;
      store_c_f32_n(acc, fvb, LAT_N, LAT_N, lane);
    }
    // h net (padded weight cols zero out the ctx part of the input)
    #pragma unroll
    for (int ntl = 0; ntl < 2; ++ntl) {
      v16h b = load_op(s.hW1 + ntl * 16 * 32, 32, lane);
      v8f acc = vzero8();
      acc = WMMA_F16F32(a, b, acc);
      float bb = s.hb1[ntl * 16 + (lane & 15)];
      #pragma unroll
      for (int i = 0; i < 8; ++i) acc[i] = softplus_f(acc[i] + bb);
      store_c_f16(acc, spb + ntl * 16, 32, lane);
    }
    lds_fence();
    {
      v16h a2 = load_op(spb, 32, lane);
      v16h b = load_op(s.hW2, 32, lane);
      v8f acc = vzero8();
      acc = WMMA_F16F32(a2, b, acc);
      float bb = s.hb2[lane & 15];
      #pragma unroll
      for (int i = 0; i < 8; ++i) acc[i] += bb;
      store_c_f32_n(acc, hvb, LAT_N, LAT_N, lane);
    }
    lds_fence();
    // elementwise g-net, log-ratio, Euler-Maruyama update
    for (int idx = lane; idx < 16 * LAT_N; idx += 32) {
      int r = idx / LAT_N, l = idx % LAT_N;
      float zv = zb[idx];
      float fv = fvb[idx], hv = hvb[idx];
      float acc = 0.f;
      for (int j = 0; j < HID_N; ++j)
        acc += softplus_f(zv * s.gW1[l * HID_N + j] + s.gb1[l * HID_N + j]) * s.gW2[l * HID_N + j];
      float gv = sigmoid_f(acc + s.gb2[l]);
      float u = (fv - hv) / gv;
      lrl += 0.5f * u * u * dtv;
      float dW = bm[((size_t)t * B_N + row0 + r) * LAT_N + l];
      float zn = zv + fv * dtv + gv * sq * dW;
      zb[idx] = zn;
      zs_ws[((size_t)(t + 1) * B_N + row0 + r) * LAT_N + l] = zn;
    }
    lds_fence();
  }
  atomicAdd(&accum[1], lrl);
}

// --------------------------------------------------------------- decoder ---

struct DecLds {
  _Float16 W1[32 * 32];    // (30,10)  -> 32x32
  _Float16 W2[32 * 32];    // (30,30)  -> 32x32
  _Float16 W3[112 * 32];   // (100,30) -> 112x32
  float b1[32], b2[32], b3[112];
  _Float16 ibuf[4][16 * 32];
  _Float16 sbuf[4][16 * 32];
  float obuf[4][16 * 112];
};

__global__ __launch_bounds__(128) void dec_kernel(
    const float* __restrict__ xs,
    const float* W1g, const float* b1g, const float* W2g, const float* b2g,
    const float* W3g, const float* b3g,
    const float* __restrict__ zs_ws, float* __restrict__ accum)
{
  __shared__ DecLds s;
  const int tid = threadIdx.x, nthr = blockDim.x;
  stage_w(s.W1, W1g, HID_N, LAT_N, 32, 32, tid, nthr);
  stage_w(s.W2, W2g, HID_N, HID_N, 32, 32, tid, nthr);
  stage_w(s.W3, W3g, 2 * OBS_N, HID_N, 112, 32, tid, nthr);
  stage_b(s.b1, b1g, HID_N, 32, tid, nthr);
  stage_b(s.b2, b2g, HID_N, 32, tid, nthr);
  stage_b(s.b3, b3g, 2 * OBS_N, 112, tid, nthr);
  __syncthreads();

  const int lane = tid & 31, w = tid >> 5;
  const size_t R0 = ((size_t)blockIdx.x * 4 + w) * 16;
  _Float16* ib = s.ibuf[w];
  _Float16* sb = s.sbuf[w];
  float* ob = s.obuf[w];

  for (int e = lane; e < 16 * 32; e += 32) {
    int r = e >> 5, c = e & 31;
    float v = (c < LAT_N) ? zs_ws[(R0 + r) * LAT_N + c] : 0.f;
    ib[e] = (_Float16)v;
  }
  lds_fence();
  {
    v16h a = load_op(ib, 32, lane);
    #pragma unroll
    for (int ntl = 0; ntl < 2; ++ntl) {
      v16h b = load_op(s.W1 + ntl * 16 * 32, 32, lane);
      v8f acc = vzero8();
      acc = WMMA_F16F32(a, b, acc);
      float bb = s.b1[ntl * 16 + (lane & 15)];
      #pragma unroll
      for (int i = 0; i < 8; ++i) acc[i] = softplus_f(acc[i] + bb);
      store_c_f16(acc, sb + ntl * 16, 32, lane);
    }
  }
  lds_fence();
  {
    v16h a = load_op(sb, 32, lane);
    #pragma unroll
    for (int ntl = 0; ntl < 2; ++ntl) {
      v16h b = load_op(s.W2 + ntl * 16 * 32, 32, lane);
      v8f acc = vzero8();
      acc = WMMA_F16F32(a, b, acc);
      float bb = s.b2[ntl * 16 + (lane & 15)];
      #pragma unroll
      for (int i = 0; i < 8; ++i) acc[i] = softplus_f(acc[i] + bb);
      store_c_f16(acc, ib + ntl * 16, 32, lane);
    }
  }
  lds_fence();
  {
    v16h a = load_op(ib, 32, lane);
    #pragma unroll
    for (int ntl = 0; ntl < 7; ++ntl) {
      v16h b = load_op(s.W3 + ntl * 16 * 32, 32, lane);
      v8f acc = vzero8();
      acc = WMMA_F16F32(a, b, acc);
      float bb = s.b3[ntl * 16 + (lane & 15)];
      #pragma unroll
      for (int i = 0; i < 8; ++i) acc[i] += bb;
      store_c_f32(acc, ob + ntl * 16, 112, lane);
    }
  }
  lds_fence();
  float lp = 0.f;
  for (int idx = lane; idx < 16 * OBS_N; idx += 32) {
    int r = idx / OBS_N, o = idx % OBS_N;
    float mean = ob[r * 112 + o];
    float lsc  = clamp_f(ob[r * 112 + OBS_N + o]);
    float x = xs[(R0 + r) * OBS_N + o];
    float d = (x - mean) * __expf(-lsc);
    lp += -0.5f * d * d - lsc - 0.5f * LOG2PI_F;
  }
  atomicAdd(&accum[2], lp);
}

// --------------------------------------------------------------- glue ------

__global__ void init_kernel(float* accum) {
  if (threadIdx.x < 8) accum[threadIdx.x] = 0.f;
}

__global__ void fin_kernel(const float* __restrict__ accum, float* __restrict__ out) {
  if (threadIdx.x == 0) {
    out[0] = accum[2] / (float)B_N;                 // log_pxs
    out[1] = (accum[0] + accum[1]) / (float)B_N;    // logqp0 + logqp_path
  }
}

extern "C" void kernel_launch(void* const* d_in, const int* in_sizes, int n_in,
                              void* d_out, int out_size, void* d_ws, size_t ws_size,
                              hipStream_t stream) {
  (void)in_sizes; (void)n_in; (void)out_size; (void)ws_size;
  const float* xs  = (const float*)d_in[0];
  const float* ts  = (const float*)d_in[1];
  const float* z0n = (const float*)d_in[2];
  const float* bm  = (const float*)d_in[3];
  const float* P[32];
  for (int i = 0; i < 32; ++i) P[i] = (const float*)d_in[4 + i];
  // params in setup_inputs insertion order:
  // 0 gru_Wih 1 gru_Whh 2 gru_bih 3 gru_bhh 4 drift_W1 5 drift_b1 6 drift_W2
  // 7 drift_b2 8 ctx_W 9 ctx_b 10 qz0_W 11 qz0_b 12 dec_W1 13 dec_b1 14 dec_W2
  // 15 dec_b2 16 dec_W3 17 dec_b3 18 f_W1 19 f_b1 20 f_W2 21 f_b2 22 h_W1
  // 23 h_b1 24 h_W2 25 h_b2 26 g_W1 27 g_b1 28 g_W2 29 g_b2 30 pz0_mean 31 pz0_logstd

  float* accum  = (float*)d_ws;
  float* ctx_ws = accum + 64;
  float* zs_ws  = ctx_ws + (size_t)T_N * B_N * CTX_N;

  init_kernel<<<1, 64, 0, stream>>>(accum);
  enc_kernel<<<B_N / 64, 128, 0, stream>>>(
      xs, ts, z0n,
      P[0], P[1], P[2], P[3], P[4], P[5], P[6], P[7],
      P[8], P[9], P[10], P[11], P[30], P[31],
      ctx_ws, zs_ws, accum);
  sde_kernel<<<B_N / 64, 128, 0, stream>>>(
      ts, bm,
      P[18], P[19], P[20], P[21], P[22], P[23], P[24], P[25],
      P[26], P[27], P[28], P[29],
      ctx_ws, zs_ws, accum);
  dec_kernel<<<(T_N * B_N) / 64, 128, 0, stream>>>(
      xs, P[12], P[13], P[14], P[15], P[16], P[17], zs_ws, accum);
  fin_kernel<<<1, 32, 0, stream>>>(accum, (float*)d_out);
}
